// CostVolumeCorrelationLayer_53309134078369
// MI455X (gfx1250) — matrix operand order repaired and checked by reading, not compiled, not run on hardware
//
#include <hip/hip_runtime.h>

typedef __attribute__((ext_vector_type(16))) _Float16 v16h;
typedef __attribute__((ext_vector_type(8)))  _Float16 v8h;
typedef __attribute__((ext_vector_type(4)))  _Float16 v4h;
typedef __attribute__((ext_vector_type(8)))  float    v8f;

#define B_    8
#define H_    128
#define W_    128
#define C_    128
#define DISP  4
#define NOFF  9                 // 2*DISP+1
#define PADW  144               // 4 left pad + 128 + 12 right pad (covers tile-1 N overrun)

// One workgroup per (b, h, dy). 8 wave32s; wave i owns pixel tile w0 = 16*i.
// out[w, dx] = (1/C) * sum_c F1[b,h,w,c] * F2[b,h+dy,w+dx,c], leaky-relu 0.1.
// Banded Gram matrix via v_wmma_f32_16x16x32_f16: A = F1 pixels (16 x 32 f16),
// B = padded F2 columns (32 x 16 f16) from LDS, 4 K-chunks, 2 N-tiles.
__global__ __launch_bounds__(256) void costvol_wmma_kernel(
    const float* __restrict__ F1, const float* __restrict__ F2,
    float* __restrict__ out)
{
    __shared__ _Float16 ldsB[PADW * C_];   // 144*128*2 = 36864 bytes

    const int b   = blockIdx.z;
    const int h   = blockIdx.y;
    const int dy  = (int)blockIdx.x - DISP;
    const int h2  = h + dy;
    const int tid = threadIdx.x;

    const long outRow   = (long)(b * H_ + h) * W_ * NOFF;   // + w*NOFF + disp
    const int  dispBase = (dy + DISP) * NOFF;

    if (h2 < 0 || h2 >= H_) {
        // entire dy-slice is zero (zero-padded F2); leaky(0)=0
        for (int i = tid; i < W_ * NOFF; i += 256) {
            int w = i / NOFF, k = i % NOFF;
            out[outRow + (long)w * NOFF + dispBase + k] = 0.0f;
        }
        return;
    }

    // ---- Stage F2[b,h2,:,:] into LDS as f16, padded left by DISP, zero edges.
    const float* f2row = F2 + (long)(b * H_ + h2) * W_ * C_;
    for (int i = tid; i < PADW * (C_ / 4); i += 256) {
        const int p  = i / (C_ / 4);         // padded column index
        const int c4 = (i % (C_ / 4)) * 4;   // channel group of 4
        const int w  = p - DISP;             // actual F2 column
        v4h v = {};
        if (w >= 0 && w < W_) {
            const float4 s = *(const float4*)(f2row + (long)w * C_ + c4);
            v[0] = (_Float16)s.x; v[1] = (_Float16)s.y;
            v[2] = (_Float16)s.z; v[3] = (_Float16)s.w;
        }
        *(v4h*)(&ldsB[p * C_ + c4]) = v;
    }
    __syncthreads();

    const int wave = tid >> 5;
    const int lane = tid & 31;
    const int half = lane >> 4;   // 0: lanes 0-15, 1: lanes 16-31
    const int lm   = lane & 15;   // A row (pixel) / B column within tile
    const int w0   = wave * 16;   // pixel tile base

    const float* f1row = F1 + ((long)(b * H_ + h) * W_ + (w0 + lm)) * C_;

    v8f acc0 = {};  // N-tile 0: padded cols w0 .. w0+15  (actual w0-4 .. w0+11)
    v8f acc1 = {};  // N-tile 1: padded cols w0+16 .. w0+31 (actual w0+12 .. w0+27)

#pragma unroll
    for (int kb = 0; kb < C_; kb += 32) {
        // A fragment (16x32 f16). CDNA5 16-bit A layout:
        //   lanes 0-15:  V0-3 = K 0..7,  V4-7 = K 16..23
        //   lanes 16-31: V0-3 = K 8..15, V4-7 = K 24..31
        v16h a;
        {
            const int cLo = kb + half * 8;        // halves 0..7
            const int cHi = kb + 16 + half * 8;   // halves 8..15
            const float4 x0 = *(const float4*)(f1row + cLo);
            const float4 x1 = *(const float4*)(f1row + cLo + 4);
            const float4 y0 = *(const float4*)(f1row + cHi);
            const float4 y1 = *(const float4*)(f1row + cHi + 4);
            const float xs[16] = { x0.x, x0.y, x0.z, x0.w,  x1.x, x1.y, x1.z, x1.w,
                                   y0.x, y0.y, y0.z, y0.w,  y1.x, y1.y, y1.z, y1.w };
#pragma unroll
            for (int i = 0; i < 16; ++i) a[i] = (_Float16)xs[i];
        }

        // B fragments (32x16 f16). Same K order as A: lane holds column
        // n = lane%16, elems 0-7 = K (lane/16)*8.., elems 8-15 = K 16+(lane/16)*8..
        const int kB = kb + half * 8;
        const v8h b0lo = *(const v8h*)(&ldsB[(w0 + lm)      * C_ + kB]);
        const v8h b0hi = *(const v8h*)(&ldsB[(w0 + lm)      * C_ + kB + 16]);
        const v8h b1lo = *(const v8h*)(&ldsB[(w0 + 16 + lm) * C_ + kB]);
        const v8h b1hi = *(const v8h*)(&ldsB[(w0 + 16 + lm) * C_ + kB + 16]);
        v16h bb0, bb1;
#pragma unroll
        for (int i = 0; i < 8; ++i) {
            bb0[i] = b0lo[i];  bb0[i + 8] = b0hi[i];
            bb1[i] = b1lo[i];  bb1[i + 8] = b1hi[i];
        }

        acc0 = __builtin_amdgcn_wmma_f32_16x16x32_f16(
            false, a, false, bb0, (short)0, acc0, false, false);
        acc1 = __builtin_amdgcn_wmma_f32_16x16x32_f16(
            false, a, false, bb1, (short)0, acc1, false, false);
    }

    // ---- Epilogue: extract dx band, mean over C, leaky-relu(0.1), scatter.
    // C/D layout: VGPR r, lane -> (M = r + 8*(lane/16), N = lane%16)
    const float scale = 1.0f / (float)C_;
#pragma unroll
    for (int r = 0; r < 8; ++r) {
        const int m = r + half * 8;       // pixel within tile
        const int w = w0 + m;
        {   // tile 0: actual col = w0 - DISP + lm  ->  dx = lm - DISP - m
            const int dx = lm - DISP - m;
            if (dx >= -DISP && dx <= DISP) {
                float v = acc0[r] * scale;
                v = (v >= 0.0f) ? v : 0.1f * v;
                out[outRow + (long)w * NOFF + dispBase + dx + DISP] = v;
            }
        }
        {   // tile 1: actual col = w0 + 16 - DISP + lm -> dx = 16 + lm - DISP - m
            const int dx = 16 + lm - DISP - m;
            if (dx >= -DISP && dx <= DISP) {
                float v = acc1[r] * scale;
                v = (v >= 0.0f) ? v : 0.1f * v;
                out[outRow + (long)w * NOFF + dispBase + dx + DISP] = v;
            }
        }
    }
}

extern "C" void kernel_launch(void* const* d_in, const int* in_sizes, int n_in,
                              void* d_out, int out_size, void* d_ws, size_t ws_size,
                              hipStream_t stream) {
    const float* F1 = (const float*)d_in[0];
    const float* F2 = (const float*)d_in[1];
    float* out = (float*)d_out;
    (void)in_sizes; (void)n_in; (void)out_size; (void)d_ws; (void)ws_size;

    dim3 grid(NOFF, H_, B_);   // (dy, h, b)
    dim3 block(256);           // 8 wave32s
    costvol_wmma_kernel<<<grid, block, 0, stream>>>(F1, F2, out);
}
